// CrossAttention_27943057228188
// MI455X (gfx1250) — matrix operand order repaired and checked by reading, not compiled, not run on hardware
//
#include <hip/hip_runtime.h>
#include <hip/hip_bf16.h>

#define N_NODES 50000
#define M_OBS   2048
#define E_DIM   512
#define O_DIM   256
#define LN_EPS  1e-6f

typedef __attribute__((ext_vector_type(16))) __bf16 bf16x16;
typedef __attribute__((ext_vector_type(8)))  float  floatx8;
typedef __attribute__((ext_vector_type(4)))  float  f32x4;

union Tile16 {
    bf16x16 v;
    f32x4   f4[2];
};

// round-to-nearest-even fp32 -> bf16 (as raw u16 bits)
__device__ __forceinline__ unsigned short f2bf(float f) {
    unsigned u = __float_as_uint(f);
    return (unsigned short)((u + 0x7FFFu + ((u >> 16) & 1u)) >> 16);
}

// ---- WMMA bf16 16x16x32 tile loaders (per ISA 7.12.2 16-bit layouts) ----
// A (16x32, MxK): lane m=lane&15; low lanes k = [0..7],[16..23]; high lanes +8.
__device__ __forceinline__ bf16x16 load_tileA(const unsigned short* base, int ld, int lane) {
    int m    = lane & 15;
    int koff = (lane & 16) ? 8 : 0;
    const unsigned short* p = base + (size_t)m * ld + koff;
    Tile16 u;
    u.f4[0] = *(const f32x4*)(p);        // k = koff .. koff+7
    u.f4[1] = *(const f32x4*)(p + 16);   // k = koff+16 .. koff+23
    return u.v;
}

// B (32x16, KxN) from row-major [J,K]: lane n=lane&15; low lanes k = [0..7],[16..23]; high lanes +8.
__device__ __forceinline__ bf16x16 load_tileB(const unsigned short* base, int ld, int lane) {
    int n    = lane & 15;
    int koff = (lane & 16) ? 8 : 0;
    const unsigned short* p = base + (size_t)n * ld + koff;
    Tile16 u;
    u.f4[0] = *(const f32x4*)(p);        // k = koff .. koff+7
    u.f4[1] = *(const f32x4*)(p + 16);   // k = koff+16 .. koff+23
    return u.v;
}

// ---- elementwise converts ----
__global__ void cvt_bf16_kernel(const float* __restrict__ in, unsigned short* __restrict__ out, int n) {
    int i = blockIdx.x * blockDim.x + threadIdx.x;
    if (i < n) out[i] = f2bf(in[i]);
}

// out[c*rows + r] = bf16(in[r*cols + c])   (transposed convert)
__global__ void cvt_bf16_transpose_kernel(const float* __restrict__ in, unsigned short* __restrict__ out,
                                          int rows, int cols) {
    int i = blockIdx.x * blockDim.x + threadIdx.x;
    if (i < rows * cols) {
        int r = i / cols, c = i % cols;
        out[(size_t)c * rows + r] = f2bf(in[i]);
    }
}

// ---- generic NT GEMM: D[i,j] = sum_k A[i,k]*B[j,k], one 16x16 tile per wave ----
__global__ void gemm_nt_bf16_kernel(const unsigned short* __restrict__ A,
                                    const unsigned short* __restrict__ B,
                                    unsigned short* __restrict__ D,
                                    int Itiles, int Jtiles, int K,
                                    int ldA, int ldB, int ldD, int transD) {
    int wid  = (blockIdx.x * blockDim.x + threadIdx.x) >> 5;   // wave-uniform
    int lane = threadIdx.x & 31;
    if (wid >= Itiles * Jtiles) return;                        // whole wave exits together
    int it = wid / Jtiles, jt = wid % Jtiles;
    const unsigned short* Ab = A + (size_t)it * 16 * ldA;
    const unsigned short* Bb = B + (size_t)jt * 16 * ldB;
    floatx8 acc = {};
    for (int k = 0; k < K; k += 32) {
        bf16x16 a = load_tileA(Ab + k, ldA, lane);
        bf16x16 b = load_tileB(Bb + k, ldB, lane);
        acc = __builtin_amdgcn_wmma_f32_16x16x32_bf16(false, a, false, b, (short)0, acc, false, false);
    }
    // C/D layout: VGPR i, low lanes -> row i, high lanes -> row i+8; col = lane&15
    int rbase = it * 16 + ((lane & 16) ? 8 : 0);
    int c     = jt * 16 + (lane & 15);
    for (int i = 0; i < 8; ++i) {
        int r = rbase + i;
        size_t idx = transD ? ((size_t)c * ldD + r) : ((size_t)r * ldD + c);
        D[idx] = f2bf(acc[i]);
    }
}

// ---- fused: out = LayerNorm(node @ C + node); C given as CT[e2*E + e1] (bf16) ----
#define LDA_PAD 520  // 16B-aligned padded row stride (ushorts) to spread LDS banks

__global__ __launch_bounds__(512) void fused_out_kernel(const float* __restrict__ node,
                                                        const unsigned short* __restrict__ CT,
                                                        const float* __restrict__ gamma,
                                                        const float* __restrict__ beta,
                                                        float* __restrict__ out) {
    __shared__ __align__(16) unsigned short sA[16 * LDA_PAD];   // 16 rows of node, bf16
    __shared__ float sOut[16 * E_DIM];                          // GEMM result, fp32

    int tid = threadIdx.x;
    const float* nrow = node + (size_t)blockIdx.x * 16 * E_DIM;

    // convert this block's 16 node rows to bf16 in LDS
    for (int i = tid; i < 16 * E_DIM; i += 512) {
        sA[(i >> 9) * LDA_PAD + (i & (E_DIM - 1))] = f2bf(nrow[i]);
    }
    __syncthreads();

    int wave = tid >> 5, lane = tid & 31;
    int jt0 = 2 * wave;                       // each of 16 waves: 2 column tiles (32 total)
    const unsigned short* B0 = CT + (size_t)(jt0 + 0) * 16 * E_DIM;
    const unsigned short* B1 = CT + (size_t)(jt0 + 1) * 16 * E_DIM;
    floatx8 acc0 = {}, acc1 = {};
    for (int k = 0; k < E_DIM; k += 32) {
        bf16x16 a  = load_tileA(sA + k, LDA_PAD, lane);          // A reused for both tiles
        bf16x16 b0 = load_tileB(B0 + k, E_DIM, lane);
        bf16x16 b1 = load_tileB(B1 + k, E_DIM, lane);
        acc0 = __builtin_amdgcn_wmma_f32_16x16x32_bf16(false, a, false, b0, (short)0, acc0, false, false);
        acc1 = __builtin_amdgcn_wmma_f32_16x16x32_bf16(false, a, false, b1, (short)0, acc1, false, false);
    }
    int rbase = (lane & 16) ? 8 : 0;
    int c0    = jt0 * 16 + (lane & 15);
    for (int i = 0; i < 8; ++i) {
        sOut[(rbase + i) * E_DIM + c0]      = acc0[i];
        sOut[(rbase + i) * E_DIM + c0 + 16] = acc1[i];
    }
    __syncthreads();

    // LayerNorm: wave w reduces row w (residual re-read from global, fp32-exact)
    int row = wave;
    const float* nres = nrow + (size_t)row * E_DIM;
    float sum = 0.f, sumsq = 0.f;
    for (int c = lane; c < E_DIM; c += 32) {
        float v = sOut[row * E_DIM + c] + nres[c];
        sum += v; sumsq += v * v;
    }
    for (int off = 16; off > 0; off >>= 1) {
        sum   += __shfl_xor(sum, off, 32);
        sumsq += __shfl_xor(sumsq, off, 32);
    }
    float mean = sum * (1.0f / E_DIM);
    float var  = sumsq * (1.0f / E_DIM) - mean * mean;
    float rs   = rsqrtf(var + LN_EPS);
    float* orow = out + ((size_t)blockIdx.x * 16 + row) * E_DIM;
    for (int c = lane; c < E_DIM; c += 32) {
        float v = sOut[row * E_DIM + c] + nres[c];
        orow[c] = (v - mean) * rs * gamma[c] + beta[c];
    }
}

extern "C" void kernel_launch(void* const* d_in, const int* in_sizes, int n_in,
                              void* d_out, int out_size, void* d_ws, size_t ws_size,
                              hipStream_t stream) {
    (void)in_sizes; (void)n_in; (void)out_size; (void)ws_size;
    const float* node  = (const float*)d_in[0];
    const float* obs   = (const float*)d_in[1];
    const float* Wq    = (const float*)d_in[2];
    const float* Wk    = (const float*)d_in[3];
    const float* Wv    = (const float*)d_in[4];
    const float* gamma = (const float*)d_in[5];
    const float* beta  = (const float*)d_in[6];
    float* out = (float*)d_out;

    char* ws = (char*)d_ws;
    size_t off = 0;
    auto alloc = [&](size_t elems) -> unsigned short* {
        unsigned short* p = (unsigned short*)(ws + off);
        off = (off + elems * sizeof(unsigned short) + 255) & ~(size_t)255;
        return p;
    };
    unsigned short* obs_bf = alloc((size_t)M_OBS * O_DIM);   // [M,O]
    unsigned short* wk_bf  = alloc((size_t)E_DIM * O_DIM);   // [E,O]
    unsigned short* wv_bf  = alloc((size_t)E_DIM * O_DIM);   // [E,O]
    unsigned short* wqT_bf = alloc((size_t)E_DIM * E_DIM);   // WqT[e1,e] = Wq[e,e1]
    unsigned short* kT_bf  = alloc((size_t)E_DIM * M_OBS);   // kT[e,m]
    unsigned short* vT_bf  = alloc((size_t)E_DIM * M_OBS);   // vT[e,m]
    unsigned short* sT_bf  = alloc((size_t)E_DIM * E_DIM);   // ST[e2,e1] = S[e1,e2], S = k^T v
    unsigned short* cT_bf  = alloc((size_t)E_DIM * E_DIM);   // CT[e2,e1] = C[e1,e2], C = Wq^T S

    // 1) precision converts
    cvt_bf16_kernel<<<(M_OBS * O_DIM + 255) / 256, 256, 0, stream>>>(obs, obs_bf, M_OBS * O_DIM);
    cvt_bf16_kernel<<<(E_DIM * O_DIM + 255) / 256, 256, 0, stream>>>(Wk, wk_bf, E_DIM * O_DIM);
    cvt_bf16_kernel<<<(E_DIM * O_DIM + 255) / 256, 256, 0, stream>>>(Wv, wv_bf, E_DIM * O_DIM);
    cvt_bf16_transpose_kernel<<<(E_DIM * E_DIM + 255) / 256, 256, 0, stream>>>(Wq, wqT_bf, E_DIM, E_DIM);

    auto gemm = [&](const unsigned short* A, const unsigned short* B, unsigned short* D,
                    int It, int Jt, int K, int ldA, int ldB, int ldD) {
        int blocks = (It * Jt * 32 + 255) / 256;  // one 16x16 tile per wave, 8 waves/block
        gemm_nt_bf16_kernel<<<blocks, 256, 0, stream>>>(A, B, D, It, Jt, K, ldA, ldB, ldD, 1);
    };
    // 2) k^T = (obs @ Wk^T)^T ; v^T = (obs @ Wv^T)^T   (stored transposed for the S GEMM)
    gemm(obs_bf, wk_bf, kT_bf, M_OBS / 16, E_DIM / 16, O_DIM, O_DIM, O_DIM, M_OBS);
    gemm(obs_bf, wv_bf, vT_bf, M_OBS / 16, E_DIM / 16, O_DIM, O_DIM, O_DIM, M_OBS);
    // 3) S^T: S[e1,e2] = sum_m k[m,e1] v[m,e2]
    gemm(kT_bf, vT_bf, sT_bf, E_DIM / 16, E_DIM / 16, M_OBS, M_OBS, M_OBS, E_DIM);
    // 4) C^T: C[e1,e2] = sum_e Wq[e,e1] S[e,e2]
    gemm(wqT_bf, sT_bf, cT_bf, E_DIM / 16, E_DIM / 16, E_DIM, E_DIM, E_DIM, E_DIM);
    // 5) fused out = LayerNorm(node @ C + node)
    fused_out_kernel<<<N_NODES / 16, 512, 0, stream>>>(node, cT_bf, gamma, beta, out);
}
